// LSTMModel_75557064671979
// MI455X (gfx1250) — matrix operand order repaired and checked by reading, not compiled, not run on hardware
//
#include <hip/hip_runtime.h>
#include <stdint.h>
#include <stddef.h>

// ---------------------------------------------------------------------------
// LSTM LM on gfx1250: bf16 WMMA everywhere, persistent pipelined scan kernel,
// LDS-staged output GEMM.   V=4096, E=512, H=1024, B=32, S=512
// ---------------------------------------------------------------------------

#define V_DIM 4096
#define E_DIM 512
#define H_DIM 1024
#define B_DIM 32
#define S_DIM 512
#define KC1   (E_DIM + H_DIM)   // 1536 : cell1 GEMM K  ([x | h1])
#define KC2   (2 * H_DIM)       // 2048 : cell2 GEMM K  ([h1 | h2])
#define MROWS (S_DIM * B_DIM)   // 16384 flattened (s,b) rows
#define SCAN_BLOCKS 32          // must all be co-resident (trivially true)

typedef __attribute__((ext_vector_type(16))) __bf16 v16bf;
typedef __attribute__((ext_vector_type(8)))  __bf16 v8bf;
typedef __attribute__((ext_vector_type(8)))  float  v8f;
typedef unsigned short ushort_t;

// ---- helpers --------------------------------------------------------------

__device__ __forceinline__ ushort_t f2bf(float x) {
  unsigned int u = __float_as_uint(x);
  u += 0x7fffu + ((u >> 16) & 1u);          // round-to-nearest-even
  return (ushort_t)(u >> 16);
}

__device__ __forceinline__ float sigmoid_f(float x) {
  return 1.0f / (1.0f + __expf(-x));
}

__device__ __forceinline__ float tanh_f(float x) {
  x = fminf(fmaxf(x, -15.0f), 15.0f);
  float e = __expf(-2.0f * x);
  return (1.0f - e) / (1.0f + e);
}

// Combine two contiguous 8-element (16B) chunks into a WMMA operand register
// block.  A operand (16x32 bf16): elems 0..15 -> K = kh*16 + i (contiguous,
// matching the B operand order).
__device__ __forceinline__ v16bf combine16(const ushort_t* p0, const ushort_t* p1) {
  v8bf lo = *reinterpret_cast<const v8bf*>(p0);
  v8bf hi = *reinterpret_cast<const v8bf*>(p1);
  v16bf r;
#pragma unroll
  for (int i = 0; i < 8; ++i) { r[i] = lo[i]; r[i + 8] = hi[i]; }
  return r;
}

__device__ __forceinline__ v16bf load_a_tile(const ushort_t* base, int half) {
  return combine16(base + half * 16, base + half * 16 + 8);
}

// B operand (32x16 bf16): lane-half holds 16 consecutive K of column n
//  = 16 contiguous elements of row n of W (since B[k][n] = W[n][k]).
__device__ __forceinline__ v16bf load_b_tile(const ushort_t* p) {
  return *reinterpret_cast<const v16bf*>(p);   // 32B-aligned global source
}

// Grid-wide barrier over SCAN_BLOCKS co-resident blocks (monotone generation).
__device__ __forceinline__ void grid_sync(int* cnt, int* gen, int nblocks, int target) {
  __syncthreads();
  if (threadIdx.x == 0) {
    __threadfence();
    int prev = __hip_atomic_fetch_add(cnt, 1, __ATOMIC_ACQ_REL, __HIP_MEMORY_SCOPE_AGENT);
    if (prev == nblocks - 1) {
      __hip_atomic_store(cnt, 0, __ATOMIC_RELAXED, __HIP_MEMORY_SCOPE_AGENT);
      __hip_atomic_store(gen, target, __ATOMIC_RELEASE, __HIP_MEMORY_SCOPE_AGENT);
    } else {
      while (__hip_atomic_load(gen, __ATOMIC_ACQUIRE, __HIP_MEMORY_SCOPE_AGENT) < target) {
        __builtin_amdgcn_s_sleep(1);
      }
    }
  }
  __syncthreads();
}

// ---- prep kernels ---------------------------------------------------------

__global__ void zero_init_kernel(unsigned int* p, int nwords) {
  for (int i = blockIdx.x * blockDim.x + threadIdx.x; i < nwords;
       i += gridDim.x * blockDim.x)
    p[i] = 0u;
}

// dst[r, 0:ca] = A[r, :];  dst[r, ca:ca+cb] = B[r, :]   (fp32 -> bf16)
__global__ void convcat_kernel(const float* __restrict__ A, const float* __restrict__ B,
                               ushort_t* __restrict__ dst, int rows, int ca, int cb) {
  int cols = ca + cb;
  size_t n = (size_t)rows * cols;
  for (size_t i = blockIdx.x * (size_t)blockDim.x + threadIdx.x; i < n;
       i += (size_t)gridDim.x * blockDim.x) {
    int r = (int)(i / cols);
    int c = (int)(i % cols);
    float v = (c < ca) ? A[(size_t)r * ca + c] : B[(size_t)r * cb + (c - ca)];
    dst[i] = f2bf(v);
  }
}

// Xbf[s*B+b, :] = bf16(emb[tokens[b, s], :])
__global__ void embed_kernel(const int* __restrict__ tokens, const float* __restrict__ emb,
                             ushort_t* __restrict__ Xbf) {
  int m = blockIdx.x;                   // s*B + b
  int s = m >> 5;
  int b = m & 31;
  int tok = tokens[b * S_DIM + s];
  const float* src = emb + (size_t)tok * E_DIM;
  ushort_t* dst = Xbf + (size_t)m * E_DIM;
  for (int e = threadIdx.x; e < E_DIM; e += blockDim.x) dst[e] = f2bf(src[e]);
}

// ---- persistent pipelined LSTM scan ---------------------------------------
// 32 blocks x 256 threads = 256 waves. Waves 0-127 (group 0) run cell 1 at
// step u; waves 128-255 (group 1) run cell 2 at step u-1 concurrently, so the
// serial critical path is one GEMM + one grid sync per timestep.
// Each wave owns (mt in {0,1} row-tile of B=32) x (jt in 0..63 tile of 16
// h-columns) and accumulates all four gates (i,f,g,o) for that tile so the
// LSTM nonlinearity is purely per-lane on the WMMA accumulator layout.
// Weights (29 MB bf16) stay L2-resident on MI455X's 192 MB L2.

__global__ __launch_bounds__(256) void lstm_scan_kernel(
    const ushort_t* __restrict__ Wc1,   // [4H, KC1] bf16  ([W_ih1 | W_hh1])
    const ushort_t* __restrict__ Wc2,   // [4H, KC2] bf16  ([W_ih2 | W_hh2])
    const float* __restrict__ b1v, const float* __restrict__ b2v,
    const ushort_t* __restrict__ Xbf,   // [MROWS, E] bf16
    ushort_t* __restrict__ h1bf,        // [2][B, H] bf16 (double buffered)
    ushort_t* __restrict__ h2bf,        // [2][B, H] bf16
    float* __restrict__ c1, float* __restrict__ c2,   // [B, H] fp32
    ushort_t* __restrict__ h2seq,       // [MROWS, H] bf16
    int* __restrict__ sync_cnt, int* __restrict__ sync_gen) {
  const int lane = threadIdx.x & 31;
  const int wib  = threadIdx.x >> 5;
  const int gw   = blockIdx.x * 8 + wib;   // 0..255
  const int grp  = gw >> 7;                // 0: cell1 chain, 1: cell2 chain
  const int task = gw & 127;
  const int mt   = task & 1;               // row tile of B=32
  const int jt   = task >> 1;              // 0..63 h-column tile
  const int half = lane >> 4;
  const int lmod = lane & 15;
  const int arow = mt * 16 + lmod;         // A-operand row (batch index)
  const int jcol = jt * 16 + lmod;         // output h column
  const int HB   = B_DIM * H_DIM;          // elements per state buffer

  for (int u = 0; u <= S_DIM; ++u) {
    if (grp == 0 && u < S_DIM) {
      // cell 1: gates = [x(u) | h1(u-1)] @ Wc1^T   (K = 1536)
      const ushort_t* hprev = h1bf + ((u + 1) & 1) * HB;   // h1(u-1)
      v8f a0 = {}, a1 = {}, a2 = {}, a3 = {};
      for (int ks = 0; ks < KC1 / 32; ++ks) {
        const ushort_t* abase =
            (ks < E_DIM / 32)
                ? (Xbf + (size_t)(u * B_DIM + arow) * E_DIM + ks * 32)
                : (hprev + (size_t)arow * H_DIM + (ks - E_DIM / 32) * 32);
        v16bf a = load_a_tile(abase, half);
        const size_t wro = (size_t)(jt * 16 + lmod) * KC1 + ks * 32 + half * 16;
        v16bf w0 = load_b_tile(Wc1 + wro);
        v16bf w1 = load_b_tile(Wc1 + (size_t)1 * H_DIM * KC1 + wro);
        v16bf w2 = load_b_tile(Wc1 + (size_t)2 * H_DIM * KC1 + wro);
        v16bf w3 = load_b_tile(Wc1 + (size_t)3 * H_DIM * KC1 + wro);
        a0 = __builtin_amdgcn_wmma_f32_16x16x32_bf16(false, a, false, w0, (short)0, a0, false, false);
        a1 = __builtin_amdgcn_wmma_f32_16x16x32_bf16(false, a, false, w1, (short)0, a1, false, false);
        a2 = __builtin_amdgcn_wmma_f32_16x16x32_bf16(false, a, false, w2, (short)0, a2, false, false);
        a3 = __builtin_amdgcn_wmma_f32_16x16x32_bf16(false, a, false, w3, (short)0, a3, false, false);
      }
      float bi = b1v[jcol], bf_ = b1v[H_DIM + jcol];
      float bg = b1v[2 * H_DIM + jcol], bo = b1v[3 * H_DIM + jcol];
      ushort_t* hout = h1bf + (u & 1) * HB;                // h1(u)
#pragma unroll
      for (int r = 0; r < 8; ++r) {
        int m = mt * 16 + half * 8 + r;                    // C/D layout row
        size_t idx = (size_t)m * H_DIM + jcol;
        float gi = sigmoid_f(a0[r] + bi);
        float gf = sigmoid_f(a1[r] + bf_);
        float gg = tanh_f(a2[r] + bg);
        float go = sigmoid_f(a3[r] + bo);
        float cn = gf * c1[idx] + gi * gg;
        c1[idx] = cn;
        hout[idx] = f2bf(go * tanh_f(cn));
      }
    } else if (grp == 1 && u >= 1) {
      // cell 2 (one step behind): gates = [h1(t) | h2(t-1)] @ Wc2^T, t = u-1
      const int t = u - 1;
      const ushort_t* h1in   = h1bf + (t & 1) * HB;        // h1(t)
      const ushort_t* h2prev = h2bf + (u & 1) * HB;        // h2(t-1)
      v8f a0 = {}, a1 = {}, a2 = {}, a3 = {};
      for (int ks = 0; ks < KC2 / 32; ++ks) {
        const ushort_t* abase =
            (ks < H_DIM / 32)
                ? (h1in + (size_t)arow * H_DIM + ks * 32)
                : (h2prev + (size_t)arow * H_DIM + (ks - H_DIM / 32) * 32);
        v16bf a = load_a_tile(abase, half);
        const size_t wro = (size_t)(jt * 16 + lmod) * KC2 + ks * 32 + half * 16;
        v16bf w0 = load_b_tile(Wc2 + wro);
        v16bf w1 = load_b_tile(Wc2 + (size_t)1 * H_DIM * KC2 + wro);
        v16bf w2 = load_b_tile(Wc2 + (size_t)2 * H_DIM * KC2 + wro);
        v16bf w3 = load_b_tile(Wc2 + (size_t)3 * H_DIM * KC2 + wro);
        a0 = __builtin_amdgcn_wmma_f32_16x16x32_bf16(false, a, false, w0, (short)0, a0, false, false);
        a1 = __builtin_amdgcn_wmma_f32_16x16x32_bf16(false, a, false, w1, (short)0, a1, false, false);
        a2 = __builtin_amdgcn_wmma_f32_16x16x32_bf16(false, a, false, w2, (short)0, a2, false, false);
        a3 = __builtin_amdgcn_wmma_f32_16x16x32_bf16(false, a, false, w3, (short)0, a3, false, false);
      }
      float bi = b2v[jcol], bf_ = b2v[H_DIM + jcol];
      float bg = b2v[2 * H_DIM + jcol], bo = b2v[3 * H_DIM + jcol];
      ushort_t* hout = h2bf + ((u + 1) & 1) * HB;          // h2(t)
#pragma unroll
      for (int r = 0; r < 8; ++r) {
        int m = mt * 16 + half * 8 + r;
        size_t idx = (size_t)m * H_DIM + jcol;
        float gi = sigmoid_f(a0[r] + bi);
        float gf = sigmoid_f(a1[r] + bf_);
        float gg = tanh_f(a2[r] + bg);
        float go = sigmoid_f(a3[r] + bo);
        float cn = gf * c2[idx] + gi * gg;
        c2[idx] = cn;
        float hv = go * tanh_f(cn);
        ushort_t hb = f2bf(hv);
        hout[idx] = hb;
        h2seq[(size_t)(t * B_DIM + m) * H_DIM + jcol] = hb;
      }
    }
    grid_sync(sync_cnt, sync_gen, SCAN_BLOCKS, u + 1);
  }
}

// ---- output head: logits[b,s,v] = h2seq[s*B+b,:] . W_out[v,:] + b_out[v] --
// Block tile 32 rows x 256 cols, K-sliced by 32, double-buffered through LDS:
//   per K-step the block stages A (32x32) + B (256x32) tiles in LDS while the
//   next slab is prefetched into registers (one __syncthreads per K-step).
// This cuts redundant L2 reads (A shared by 4 waves, B by 2) ~2.2x vs direct
// streaming.  LDS rows padded to 40 bf16 (80B) so every LDS access is 16B
// aligned and bank-spread.  8 waves: mt = w&1 (16-row tile), ng = w>>1
// (64-col group); each wave holds 4 accumulators (16x64 tile).

#define LPAD 40   // padded LDS row stride in bf16 elements

__global__ __launch_bounds__(256) void logits_kernel(
    const ushort_t* __restrict__ h2seq,   // [MROWS, H] bf16
    const ushort_t* __restrict__ Woutbf,  // [V, H] bf16
    const float* __restrict__ b_out, float* __restrict__ out) {
  __shared__ ushort_t Alds[2][32 * LPAD];     //  5 KB
  __shared__ ushort_t Blds[2][256 * LPAD];    // 40 KB

  const int t    = threadIdx.x;
  const int lane = t & 31;
  const int wib  = t >> 5;
  const int half = lane >> 4;
  const int lmod = lane & 15;
  const int mt   = wib & 1;
  const int ng   = wib >> 1;
  const int brow = blockIdx.x & 511;      // 512 row blocks of 32
  const int bcol = blockIdx.x >> 9;       // 16 col blocks of 256
  const int mbase = brow * 32 + mt * 16;
  const int vbase = bcol * 256 + ng * 64;

  // global source pointers for this thread's staging work
  const ushort_t* bsrc = Woutbf + (size_t)(bcol * 256 + t) * H_DIM;       // row t of B tile
  const ushort_t* asrc = h2seq + (size_t)(brow * 32 + (t >> 2)) * H_DIM + (t & 3) * 8;

  uint4 pb0, pb1, pb2, pb3, pa;
  auto load_slab = [&](int kb) {
    const uint4* p = reinterpret_cast<const uint4*>(bsrc + kb * 32);
    pb0 = p[0]; pb1 = p[1]; pb2 = p[2]; pb3 = p[3];
    if (t < 128) pa = *reinterpret_cast<const uint4*>(asrc + kb * 32);
  };
  auto store_slab = [&](int p) {
    uint4* bd = reinterpret_cast<uint4*>(&Blds[p][t * LPAD]);             // t*80B, 16B aligned
    bd[0] = pb0; bd[1] = pb1; bd[2] = pb2; bd[3] = pb3;
    if (t < 128)
      *reinterpret_cast<uint4*>(&Alds[p][(t >> 2) * LPAD + (t & 3) * 8]) = pa;
  };

  v8f a0 = {}, a1 = {}, a2 = {}, a3 = {};
  load_slab(0);
  int p = 0;
  for (int kb = 0; kb < H_DIM / 32; ++kb) {
    store_slab(p);
    __syncthreads();
    if (kb + 1 < H_DIM / 32) load_slab(kb + 1);
    const ushort_t* ab = &Alds[p][(mt * 16 + lmod) * LPAD];
    v16bf a = load_a_tile(ab, half);
    const ushort_t* bb = &Blds[p][(ng * 64 + lmod) * LPAD + half * 16];
    v16bf w0 = combine16(bb + 0 * 16 * LPAD, bb + 0 * 16 * LPAD + 8);
    v16bf w1 = combine16(bb + 1 * 16 * LPAD, bb + 1 * 16 * LPAD + 8);
    v16bf w2 = combine16(bb + 2 * 16 * LPAD, bb + 2 * 16 * LPAD + 8);
    v16bf w3 = combine16(bb + 3 * 16 * LPAD, bb + 3 * 16 * LPAD + 8);
    a0 = __builtin_amdgcn_wmma_f32_16x16x32_bf16(false, a, false, w0, (short)0, a0, false, false);
    a1 = __builtin_amdgcn_wmma_f32_16x16x32_bf16(false, a, false, w1, (short)0, a1, false, false);
    a2 = __builtin_amdgcn_wmma_f32_16x16x32_bf16(false, a, false, w2, (short)0, a2, false, false);
    a3 = __builtin_amdgcn_wmma_f32_16x16x32_bf16(false, a, false, w3, (short)0, a3, false, false);
    p ^= 1;
  }

  float bo0 = b_out[vbase + 0 * 16 + lmod];
  float bo1 = b_out[vbase + 1 * 16 + lmod];
  float bo2 = b_out[vbase + 2 * 16 + lmod];
  float bo3 = b_out[vbase + 3 * 16 + lmod];
#pragma unroll
  for (int r = 0; r < 8; ++r) {
    int m = mbase + half * 8 + r;          // flattened (s*B + b)
    int bb_ = m & 31;
    int ss = m >> 5;
    size_t obase = ((size_t)bb_ * S_DIM + ss) * V_DIM + vbase;
    out[obase + 0 * 16 + lmod] = a0[r] + bo0;
    out[obase + 1 * 16 + lmod] = a1[r] + bo1;
    out[obase + 2 * 16 + lmod] = a2[r] + bo2;
    out[obase + 3 * 16 + lmod] = a3[r] + bo3;
  }
}

// ---- host-side orchestration ----------------------------------------------

extern "C" void kernel_launch(void* const* d_in, const int* in_sizes, int n_in,
                              void* d_out, int out_size, void* d_ws, size_t ws_size,
                              hipStream_t stream) {
  (void)in_sizes; (void)n_in; (void)out_size;
  const int*   tokens = (const int*)d_in[0];
  const float* emb    = (const float*)d_in[1];
  const float* W_ih1  = (const float*)d_in[2];
  const float* W_hh1  = (const float*)d_in[3];
  const float* b1     = (const float*)d_in[4];
  const float* W_ih2  = (const float*)d_in[5];
  const float* W_hh2  = (const float*)d_in[6];
  const float* b2     = (const float*)d_in[7];
  const float* W_out  = (const float*)d_in[8];
  const float* b_out  = (const float*)d_in[9];
  float* out = (float*)d_out;

  char* ws = (char*)d_ws;
  size_t off = 0;
  auto take = [&](size_t bytes) {
    size_t o = off;
    off = (off + bytes + 255) & ~(size_t)255;
    return o;
  };
  const size_t off_Wc1   = take((size_t)4 * H_DIM * KC1 * 2);       // 12.6 MB
  const size_t off_Wc2   = take((size_t)4 * H_DIM * KC2 * 2);       // 16.8 MB
  const size_t off_Wout  = take((size_t)V_DIM * H_DIM * 2);         //  8.4 MB
  const size_t off_Xbf   = take((size_t)MROWS * E_DIM * 2);         // 16.8 MB
  const size_t off_h2seq = take((size_t)MROWS * H_DIM * 2);         // 33.6 MB
  const size_t off_state = off;                                     // zeroed
  const size_t off_h1bf  = take((size_t)2 * B_DIM * H_DIM * 2);
  const size_t off_h2bf  = take((size_t)2 * B_DIM * H_DIM * 2);
  const size_t off_c1    = take((size_t)B_DIM * H_DIM * 4);
  const size_t off_c2    = take((size_t)B_DIM * H_DIM * 4);
  const size_t off_sync  = take(256);
  const size_t state_bytes = off - off_state;
  if (ws_size < off) return;   // scratch too small; nothing safe to do

  ushort_t* Wc1    = (ushort_t*)(ws + off_Wc1);
  ushort_t* Wc2    = (ushort_t*)(ws + off_Wc2);
  ushort_t* Woutbf = (ushort_t*)(ws + off_Wout);
  ushort_t* Xbf    = (ushort_t*)(ws + off_Xbf);
  ushort_t* h2seq  = (ushort_t*)(ws + off_h2seq);
  ushort_t* h1bf   = (ushort_t*)(ws + off_h1bf);
  ushort_t* h2bf   = (ushort_t*)(ws + off_h2bf);
  float*    c1     = (float*)(ws + off_c1);
  float*    c2     = (float*)(ws + off_c2);
  int*      sync_cnt = (int*)(ws + off_sync);
  int*      sync_gen = sync_cnt + 1;

  const int nwords = (int)(state_bytes / 4);
  zero_init_kernel<<<(nwords + 255) / 256, 256, 0, stream>>>(
      (unsigned int*)(ws + off_state), nwords);
  convcat_kernel<<<2048, 256, 0, stream>>>(W_ih1, W_hh1, Wc1, 4 * H_DIM, E_DIM, H_DIM);
  convcat_kernel<<<2048, 256, 0, stream>>>(W_ih2, W_hh2, Wc2, 4 * H_DIM, H_DIM, H_DIM);
  convcat_kernel<<<2048, 256, 0, stream>>>(W_out, W_out, Woutbf, V_DIM, H_DIM, 0);
  embed_kernel<<<MROWS, 256, 0, stream>>>(tokens, emb, Xbf);
  lstm_scan_kernel<<<SCAN_BLOCKS, 256, 0, stream>>>(
      Wc1, Wc2, b1, b2, Xbf, h1bf, h2bf, c1, c2, h2seq, sync_cnt, sync_gen);
  logits_kernel<<<512 * 16, 256, 0, stream>>>(h2seq, Woutbf, b_out, out);
}
